// AtomUpdateBlock_9388798509342
// MI455X (gfx1250) — matrix operand order repaired and checked by reading, not compiled, not run on hardware
//
#include <hip/hip_runtime.h>

#define DE 256
#define DR 16
#define EPB 64     // edges per block (edge kernel)
#define MT  64     // atom rows per block (mlp kernel)
#define SP  264    // padded LDS row stride (floats)

typedef __attribute__((ext_vector_type(16))) __bf16 v16bf;
typedef __attribute__((ext_vector_type(8)))  float  v8f;

__device__ __forceinline__ v8f wmma_bf16(v16bf a, v16bf b, v8f c) {
  // D = A(16x32 bf16) * B(32x16 bf16) + C(16x16 f32)
  return __builtin_amdgcn_wmma_f32_16x16x32_bf16(
      /*neg_a=*/false, a, /*neg_b=*/false, b,
      /*c_mod=*/(short)0, c, /*reuse_a=*/false, /*reuse_b=*/false);
}

// Split 16 f32 into hi/lo bf16 fragments (bf16x3 error compensation).
__device__ __forceinline__ void split16(const float* v, v16bf& hi, v16bf& lo) {
#pragma unroll
  for (int j = 0; j < 16; ++j) {
    float f = v[j];
    __bf16 h = (__bf16)f;
    hi[j] = h;
    lo[j] = (__bf16)(f - (float)h);
  }
}

// ---------------------------------------------------------------------------
// Kernel 1: zero the segment-sum accumulator (d_out doubles as x2 buffer).
// ---------------------------------------------------------------------------
__global__ void k_zero(float* __restrict__ p, long long n4) {
  long long i = (long long)blockIdx.x * blockDim.x + threadIdx.x;
  if (i < n4) {
    float4 z = {0.f, 0.f, 0.f, 0.f};
    *(float4*)(p + i * 4) = z;
  }
}

// ---------------------------------------------------------------------------
// Kernel 2: split weights into bf16 hi/lo.
//  - W1 + res_W  -> [layer][n][k]           (k contiguous per output column)
//  - W_rbf       -> [n][32], k=16..31 zero  (K padded for the 16x16x32 WMMA)
// ---------------------------------------------------------------------------
__global__ void k_split_w(const float* __restrict__ W1,
                          const float* __restrict__ resW,
                          const float* __restrict__ Wrbf,
                          unsigned short* __restrict__ whi_us,
                          unsigned short* __restrict__ wlo_us,
                          unsigned short* __restrict__ rhi_us,
                          unsigned short* __restrict__ rlo_us,
                          int nLayers) {
  int idx = blockIdx.x * blockDim.x + threadIdx.x;
  int nMain = nLayers * DE * DE;
  if (idx < nMain) {
    int l = idx >> 16;          // 65536 = 256*256
    int r = idx & 65535;
    int n = r & 255;
    int k = r >> 8;
    float w = (l == 0) ? W1[k * DE + n] : resW[(size_t)(l - 1) * 65536 + k * DE + n];
    __bf16 h = (__bf16)w;
    __bf16 lo = (__bf16)(w - (float)h);
    ((__bf16*)whi_us)[(size_t)l * 65536 + n * DE + k] = h;
    ((__bf16*)wlo_us)[(size_t)l * 65536 + n * DE + k] = lo;
  } else if (idx < nMain + DE * 32) {
    int r = idx - nMain;
    int n = r >> 5;
    int k = r & 31;
    float w = (k < DR) ? Wrbf[k * DE + n] : 0.f;
    __bf16 h = (__bf16)w;
    __bf16 lo = (__bf16)(w - (float)h);
    ((__bf16*)rhi_us)[n * 32 + k] = h;
    ((__bf16*)rlo_us)[n * 32 + k] = lo;
  }
}

// ---------------------------------------------------------------------------
// Kernel 3: fused edge gate + scatter.
//   gate = rbf @ W_rbf  (WMMA, K padded 16->32), x = m * gate,
//   atomicAdd into x2[id_j].  256 threads = 8 waves; 64 edges per block.
// ---------------------------------------------------------------------------
__global__ __launch_bounds__(256) void k_edge(const float* __restrict__ m,
                                              const float* __restrict__ rbf,
                                              const int* __restrict__ id_j,
                                              const unsigned short* __restrict__ rhi_us,
                                              const unsigned short* __restrict__ rlo_us,
                                              float* __restrict__ x2,
                                              int nEdges) {
  __shared__ float s_rbf[EPB * 17];     // rbf tile, padded
  __shared__ int   s_id[EPB];
  __shared__ float s_gate[EPB * 260];   // gate tile f32, 66.6 KB

  const int tid = threadIdx.x;
  const int e0  = blockIdx.x * EPB;
  const __bf16* rhi = (const __bf16*)rhi_us;
  const __bf16* rlo = (const __bf16*)rlo_us;

  for (int i = tid; i < EPB * DR; i += 256) {
    int e = i >> 4, k = i & 15;
    float v = (e0 + e < nEdges) ? rbf[(size_t)(e0 + e) * DR + k] : 0.f;
    s_rbf[e * 17 + k] = v;
  }
  if (tid < EPB) s_id[tid] = (e0 + tid < nEdges) ? id_j[e0 + tid] : 0;
  __syncthreads();

  // ---- WMMA phase: 8 waves x 8 col tiles; 4 row strips x 16 col tiles ----
  const int lane  = tid & 31;
  const int wv    = tid >> 5;
  const int strip = wv & 3;       // which 16-edge strip
  const int ch    = wv >> 2;      // which half of the 256 columns
  const int ln    = lane & 15;
  const bool lowh = (lane < 16);
  const int kb0   = lowh ? 0 : 8;   // unused; A element s of lane half h is K 16h+s, as in B
  const int kbB   = lowh ? 0 : 16;  // B K base (k>=16 rows are stored zeros)
  const int mrow  = strip * 16 + ln;

  v8f acc[8] = {};

  // A fragment (shared by all 8 col tiles): lanes<16 K 0..15 real, lanes>=16 zero pad.
  v16bf ahi = {}, alo = {};
#pragma unroll
  for (int j = 0; j < 16; ++j) {
    float f = lowh ? s_rbf[mrow * 17 + j] : 0.f;
    __bf16 h = (__bf16)f;
    ahi[j] = h;
    alo[j] = (__bf16)(f - (float)h);
  }

#pragma unroll
  for (int t = 0; t < 8; ++t) {
    int n = (ch * 8 + t) * 16 + ln;
    // Unconditional 32B loads from the L2-resident padded table.
    v16bf bhi = *(const v16bf*)(rhi + n * 32 + kbB);
    v16bf blo = *(const v16bf*)(rlo + n * 32 + kbB);
    acc[t] = wmma_bf16(ahi, bhi, acc[t]);
    acc[t] = wmma_bf16(alo, bhi, acc[t]);
    acc[t] = wmma_bf16(ahi, blo, acc[t]);
  }

  // Store gate tiles to LDS (C layout: vgpr v -> M=v (lanes<16) / v+8)
  const int mhalf = lowh ? 0 : 8;
#pragma unroll
  for (int t = 0; t < 8; ++t) {
    int col = (ch * 8 + t) * 16 + ln;
#pragma unroll
    for (int v = 0; v < 8; ++v) {
      int row = strip * 16 + v + mhalf;
      s_gate[row * 260 + col] = acc[t][v];
    }
  }
  __syncthreads();

  // ---- elementwise + scatter phase: coalesced m, f32 atomics into L2 ----
  int nv = nEdges - e0;
  if (nv >= EPB) {
#pragma unroll 4
    for (int e = 0; e < EPB; ++e) {
      float val = m[(size_t)(e0 + e) * DE + tid] * s_gate[e * 260 + tid];
      unsafeAtomicAdd(&x2[(size_t)s_id[e] * DE + tid], val);
    }
  } else {
    for (int e = 0; e < nv; ++e) {
      float val = m[(size_t)(e0 + e) * DE + tid] * s_gate[e * 260 + tid];
      unsafeAtomicAdd(&x2[(size_t)s_id[e] * DE + tid], val);
    }
  }
}

// ---------------------------------------------------------------------------
// Kernel 4: atom MLP. One block owns a 64-row tile through all 7 layers.
// Activations live in LDS (f32); GEMMs via bf16x3 WMMA; weights from L2.
// ---------------------------------------------------------------------------
__global__ __launch_bounds__(256) void k_mlp(const float* __restrict__ x2,
                                             const unsigned short* __restrict__ whi_us,
                                             const unsigned short* __restrict__ wlo_us,
                                             const float* __restrict__ scale_p,
                                             float* __restrict__ out,
                                             int nAtoms, int nLayers) {
  __shared__ float b0[MT * SP];   // residual / primary activation
  __shared__ float b1[MT * SP];   // pair intermediate

  const int tid = threadIdx.x;
  const int rowBase = blockIdx.x * MT;
  int nv = nAtoms - rowBase; if (nv > MT) nv = MT;
  const float scale = scale_p[0];
  const __bf16* whi = (const __bf16*)whi_us;
  const __bf16* wlo = (const __bf16*)wlo_us;

  // Load x2 tile * scale into b0 (zeros past the tail).
#pragma unroll
  for (int it = 0; it < 16; ++it) {
    int row = it * 4 + (tid >> 6);
    int c4  = (tid & 63) * 4;
    float4 v = {0.f, 0.f, 0.f, 0.f};
    if (row < nv) v = *(const float4*)(x2 + (size_t)(rowBase + row) * DE + c4);
    v.x *= scale; v.y *= scale; v.z *= scale; v.w *= scale;
    *(float4*)(&b0[row * SP + c4]) = v;
  }
  __syncthreads();

  const int lane  = tid & 31;
  const int wv    = tid >> 5;
  const int strip = wv & 3;
  const int ch    = wv >> 2;
  const int ln    = lane & 15;
  const bool lowh = (lane < 16);
  const int kb0   = lowh ? 0 : 16;   // A K base
  const int kbB   = lowh ? 0 : 16;   // B K base
  const int mhalf = lowh ? 0 : 8;
  const float kSilu = 1.6666666666666667f;  // 1/0.6
  const float kInvS2 = 0.7071067811865476f;

  for (int layer = 0; layer < nLayers; ++layer) {
    const bool isFirst = (layer == 0);
    const bool isA = (!isFirst) && (layer & 1);     // first layer of a pair
    const bool isLast = (layer == nLayers - 1);
    const float* in = (isFirst || isA) ? b0 : b1;
    const __bf16* WH = whi + (size_t)layer * 65536;
    const __bf16* WL = wlo + (size_t)layer * 65536;

    v8f acc[8] = {};
#pragma unroll
    for (int kt = 0; kt < 8; ++kt) {
      // A fragment from LDS: elements 0..15 -> K kt*32+kb0..kb0+15
      const float* ap = in + (strip * 16 + ln) * SP + kt * 32 + kb0;
      float av[16];
      *(float4*)(av + 0)  = *(const float4*)(ap + 0);
      *(float4*)(av + 4)  = *(const float4*)(ap + 4);
      *(float4*)(av + 8)  = *(const float4*)(ap + 8);
      *(float4*)(av + 12) = *(const float4*)(ap + 12);
      v16bf ahi, alo;
      split16(av, ahi, alo);
#pragma unroll
      for (int t = 0; t < 8; ++t) {
        size_t woff = (size_t)((ch * 8 + t) * 16 + ln) * DE + kt * 32 + kbB;
        v16bf bhi = *(const v16bf*)(WH + woff);
        v16bf blo = *(const v16bf*)(WL + woff);
        acc[t] = wmma_bf16(ahi, bhi, acc[t]);
        acc[t] = wmma_bf16(alo, bhi, acc[t]);
        acc[t] = wmma_bf16(ahi, blo, acc[t]);
      }
    }

    // ScaledSiLU
#pragma unroll
    for (int t = 0; t < 8; ++t) {
#pragma unroll
      for (int v = 0; v < 8; ++v) {
        float x = acc[t][v];
        acc[t][v] = x * kSilu / (1.f + __expf(-x));
      }
    }

    __syncthreads();   // all A reads of `in` complete before stores

    if (isLast) {
#pragma unroll
      for (int t = 0; t < 8; ++t) {
        int col = (ch * 8 + t) * 16 + ln;
#pragma unroll
        for (int v = 0; v < 8; ++v) {
          int row = strip * 16 + v + mhalf;
          float r = b0[row * SP + col];
          if (rowBase + row < nAtoms)
            out[(size_t)(rowBase + row) * DE + col] = (r + acc[t][v]) * kInvS2;
        }
      }
    } else if (isFirst) {
#pragma unroll
      for (int t = 0; t < 8; ++t) {
        int col = (ch * 8 + t) * 16 + ln;
#pragma unroll
        for (int v = 0; v < 8; ++v)
          b0[(strip * 16 + v + mhalf) * SP + col] = acc[t][v];
      }
    } else if (isA) {
#pragma unroll
      for (int t = 0; t < 8; ++t) {
        int col = (ch * 8 + t) * 16 + ln;
#pragma unroll
        for (int v = 0; v < 8; ++v)
          b1[(strip * 16 + v + mhalf) * SP + col] = acc[t][v];
      }
    } else {  // second layer of a pair: residual into b0
#pragma unroll
      for (int t = 0; t < 8; ++t) {
        int col = (ch * 8 + t) * 16 + ln;
#pragma unroll
        for (int v = 0; v < 8; ++v) {
          int row = strip * 16 + v + mhalf;
          b0[row * SP + col] = (b0[row * SP + col] + acc[t][v]) * kInvS2;
        }
      }
    }
    __syncthreads();
  }
}

// ---------------------------------------------------------------------------
extern "C" void kernel_launch(void* const* d_in, const int* in_sizes, int n_in,
                              void* d_out, int out_size, void* d_ws, size_t ws_size,
                              hipStream_t stream) {
  const float* m     = (const float*)d_in[1];
  const float* rbf   = (const float*)d_in[2];
  const int*   id_j  = (const int*)d_in[3];
  const float* W_rbf = (const float*)d_in[4];
  const float* W1    = (const float*)d_in[5];
  const float* res_W = (const float*)d_in[6];
  const float* scale = (const float*)d_in[7];

  const int nAtoms = in_sizes[0] / DE;
  const int nEdges = in_sizes[1] / DE;
  const int nRes   = in_sizes[6] / (2 * DE * DE);   // 3
  const int nLayers = 1 + 2 * nRes;                 // 7

  float* out = (float*)d_out;
  unsigned short* whi = (unsigned short*)d_ws;                 // nLayers*65536 bf16
  unsigned short* wlo = whi + (size_t)nLayers * 65536;         // nLayers*65536 bf16
  unsigned short* rhi = wlo + (size_t)nLayers * 65536;         // 256*32 bf16 (padded)
  unsigned short* rlo = rhi + (size_t)DE * 32;                 // 256*32 bf16 (padded)

  // 1) zero segment-sum accumulator (d_out doubles as x2)
  long long n4 = (long long)nAtoms * DE / 4;
  k_zero<<<(unsigned)((n4 + 255) / 256), 256, 0, stream>>>(out, n4);

  // 2) split weights to bf16 hi/lo (transposed; W_rbf K-padded to 32)
  int wElems = nLayers * DE * DE + DE * 32;
  k_split_w<<<(wElems + 255) / 256, 256, 0, stream>>>(W1, res_W, W_rbf,
                                                      whi, wlo, rhi, rlo, nLayers);

  // 3) fused edge gate + scatter-add
  k_edge<<<(nEdges + EPB - 1) / EPB, 256, 0, stream>>>(m, rbf, id_j, rhi, rlo,
                                                       out, nEdges);

  // 4) atom MLP, in-place on d_out
  k_mlp<<<(nAtoms + MT - 1) / MT, 256, 0, stream>>>(out, whi, wlo, scale, out,
                                                    nAtoms, nLayers);
}
